// LeakyDecisionAccumulator_23210003268375
// MI455X (gfx1250) — matrix operand order repaired and checked by reading, not compiled, not run on hardware
//
#include <hip/hip_runtime.h>

typedef __bf16 bf16_t;
typedef __attribute__((ext_vector_type(16))) bf16_t v16bf;
typedef __attribute__((ext_vector_type(8)))  float  v8f;

#define D_DIM 512
#define B_DIM 64
#define T_DIM 512
#define M_TOT (B_DIM * T_DIM)   // 32768
#define SCAN_BLOCKS 16

__device__ __forceinline__ unsigned short f2bf(float x) {
  // round-to-nearest-even f32 -> bf16
  unsigned int u = __float_as_uint(x);
  u += 0x7FFFu + ((u >> 16) & 1u);
  return (unsigned short)(u >> 16);
}
__device__ __forceinline__ float bf2f(unsigned short h) {
  return __uint_as_float(((unsigned int)h) << 16);
}

union Frag {
  uint4 q[2];
  v16bf v;
};

// ---------------------------------------------------------------------------
// Evidence f32 -> bf16 (row-major, layout preserved). 4 elements/thread.
// grid = 16384 x 256
// ---------------------------------------------------------------------------
__global__ void evconv_kernel(const float* __restrict__ ev,
                              unsigned short* __restrict__ evbf) {
  size_t i = ((size_t)blockIdx.x * blockDim.x + threadIdx.x) * 4;
  float4 x = *(const float4*)(ev + i);
  unsigned long long p =
      (unsigned long long)((unsigned)f2bf(x.x) | ((unsigned)f2bf(x.y) << 16)) |
      ((unsigned long long)((unsigned)f2bf(x.z) | ((unsigned)f2bf(x.w) << 16)) << 32);
  *(unsigned long long*)(evbf + i) = p;
}

// ---------------------------------------------------------------------------
// Prep: WT[n][k] = bf16(W[k][n]) for 4 weights; zero belief ping + barrier.
// grid = 1024 x 256
// ---------------------------------------------------------------------------
__global__ void prep_kernel(const float* __restrict__ We, const float* __restrict__ Wp,
                            const float* __restrict__ Wl, const float* __restrict__ Ww,
                            unsigned short* __restrict__ WeT, unsigned short* __restrict__ WpT,
                            unsigned short* __restrict__ WlT, unsigned short* __restrict__ WwT,
                            unsigned short* __restrict__ belief0,
                            unsigned int* __restrict__ bar) {
  int i = blockIdx.x * blockDim.x + threadIdx.x;   // i = n*512 + k
  int n = i >> 9;
  int k = i & 511;
  int src = k * D_DIM + n;
  WeT[i] = f2bf(We[src]);
  WpT[i] = f2bf(Wp[src]);
  WlT[i] = f2bf(Wl[src]);
  WwT[i] = f2bf(Ww[src]);
  if (i < B_DIM * D_DIM) belief0[i] = 0;           // bf16 zero
  if (i < 2) bar[i] = 0u;
}

// ---------------------------------------------------------------------------
// Phase 1: fused triple GEMM (bf16 x bf16 -> f32 accum) over the sequence.
//   ev_proj -> d_out (f32), leak/write -> ws (bf16, sigmoid applied)
// grid = (2048, 4), block = 256 (8 wave32 waves); one 16x16 tile per wave.
// ---------------------------------------------------------------------------
__global__ void proj_kernel(const unsigned short* __restrict__ evbf,
                            const unsigned short* __restrict__ WeT,
                            const unsigned short* __restrict__ WlT,
                            const unsigned short* __restrict__ WwT,
                            const float* __restrict__ be,
                            const float* __restrict__ bl,
                            const float* __restrict__ bw,
                            float* __restrict__ evproj,
                            unsigned short* __restrict__ leak,
                            unsigned short* __restrict__ wrt) {
  const int lane = threadIdx.x & 31;
  const int w    = threadIdx.x >> 5;
  const int row0 = blockIdx.x * 16;
  const int col0 = (blockIdx.y * 8 + w) * 16;
  const int g    = lane >> 4;
  const int ln   = lane & 15;

  v8f accE = {};
  v8f accL = {};
  v8f accW = {};

  const bf16_t* arow  = (const bf16_t*)evbf + (size_t)(row0 + ln) * D_DIM;
  const bf16_t* bErow = (const bf16_t*)WeT  + (size_t)(col0 + ln) * D_DIM;
  const bf16_t* bLrow = (const bf16_t*)WlT  + (size_t)(col0 + ln) * D_DIM;
  const bf16_t* bWrow = (const bf16_t*)WwT  + (size_t)(col0 + ln) * D_DIM;

  for (int kt = 0; kt < D_DIM; kt += 32) {
    const int ka = kt + 16 * g;
    Frag a;
    a.q[0] = *(const uint4*)(arow + ka);
    a.q[1] = *(const uint4*)(arow + ka + 8);

    const int kb = kt + 16 * g;
    v16bf bE = *(const v16bf*)(bErow + kb);
    v16bf bL = *(const v16bf*)(bLrow + kb);
    v16bf bW = *(const v16bf*)(bWrow + kb);

    accE = __builtin_amdgcn_wmma_f32_16x16x32_bf16(false, a.v, false, bE, (short)0, accE, false, false);
    accL = __builtin_amdgcn_wmma_f32_16x16x32_bf16(false, a.v, false, bL, (short)0, accL, false, false);
    accW = __builtin_amdgcn_wmma_f32_16x16x32_bf16(false, a.v, false, bW, (short)0, accW, false, false);
  }

  const int col = col0 + ln;
  const float vbe = be[col], vbl = bl[col], vbw = bw[col];
#pragma unroll
  for (int v = 0; v < 8; ++v) {
    const int row = row0 + v + 8 * g;
    const size_t idx = (size_t)row * D_DIM + col;
    evproj[idx] = accE[v] + vbe;
    const float l  = accL[v] + vbl;
    const float wv = accW[v] + vbw;
    leak[idx] = f2bf(1.0f / (1.0f + __expf(-l)));
    wrt[idx]  = f2bf(1.0f / (1.0f + __expf(-wv)));
  }
}

// ---------------------------------------------------------------------------
// L2-resident sense-reversing grid barrier (all SCAN_BLOCKS co-resident).
// bar[0] = arrival count, bar[1] = generation.
// ---------------------------------------------------------------------------
__device__ __forceinline__ void grid_barrier(unsigned int* bar) {
  __syncthreads();
  if (threadIdx.x == 0) {
    __threadfence();
    unsigned int gen = __hip_atomic_load(&bar[1], __ATOMIC_RELAXED,
                                         __HIP_MEMORY_SCOPE_AGENT);
    unsigned int a = __hip_atomic_fetch_add(&bar[0], 1u, __ATOMIC_ACQ_REL,
                                            __HIP_MEMORY_SCOPE_AGENT);
    if (a == SCAN_BLOCKS - 1) {
      __hip_atomic_store(&bar[0], 0u, __ATOMIC_RELAXED, __HIP_MEMORY_SCOPE_AGENT);
      __hip_atomic_fetch_add(&bar[1], 1u, __ATOMIC_RELEASE, __HIP_MEMORY_SCOPE_AGENT);
    } else {
      while (__hip_atomic_load(&bar[1], __ATOMIC_ACQUIRE,
                               __HIP_MEMORY_SCOPE_AGENT) == gen) {
        __builtin_amdgcn_s_sleep(2);
      }
    }
    __threadfence();
  }
  __syncthreads();
}

// ---------------------------------------------------------------------------
// Phase 2: whole recurrence in ONE persistent kernel.
//   cand   = tanh(prevB @ Wp + bp + ev_proj[:,t,:])
//   belief = leak*prevB_f32 + write*cand
// prevB (A matrix) ping-pongs as bf16; prevB_f32 read from d_out[:,t-1,:]
// (written by this same thread last step). grid = 16 x 256, 1 tile/wave.
// ---------------------------------------------------------------------------
__global__ void scan_kernel(const unsigned short* __restrict__ WpT,
                            const float* __restrict__ bp,
                            const unsigned short* __restrict__ leak,
                            const unsigned short* __restrict__ wrt,
                            float* __restrict__ outbuf,
                            unsigned short* __restrict__ bel0,
                            unsigned short* __restrict__ bel1,
                            unsigned int* __restrict__ bar) {
  const int lane = threadIdx.x & 31;
  const int w    = threadIdx.x >> 5;
  const int tile = blockIdx.x * 8 + w;       // 0..127
  const int row0 = (tile >> 5) * 16;         // batch tile (0..3)
  const int col0 = (tile & 31) * 16;         // feature tile (0..31)
  const int g    = lane >> 4;
  const int ln   = lane & 15;
  const int col  = col0 + ln;

  const bf16_t* brow = (const bf16_t*)WpT + (size_t)col * D_DIM;
  const float   vbp  = bp[col];

#pragma unroll 1
  for (int t = 0; t < T_DIM; ++t) {
    const bf16_t* cur = (const bf16_t*)((t & 1) ? bel1 : bel0);
    unsigned short* nxt = (t & 1) ? bel0 : bel1;

    v8f acc = {};
    const bf16_t* arow = cur + (size_t)(row0 + ln) * D_DIM;
#pragma unroll
    for (int kt = 0; kt < D_DIM; kt += 32) {
      const int ka = kt + 16 * g;
      Frag a;
      a.q[0] = *(const uint4*)(arow + ka);
      a.q[1] = *(const uint4*)(arow + ka + 8);
      v16bf bb = *(const v16bf*)(brow + kt + 16 * g);
      acc = __builtin_amdgcn_wmma_f32_16x16x32_bf16(false, a.v, false, bb, (short)0, acc, false, false);
    }

#pragma unroll
    for (int v = 0; v < 8; ++v) {
      const int b = row0 + v + 8 * g;
      const size_t gidx = ((size_t)b * T_DIM + (size_t)t) * D_DIM + col;
      const float x    = acc[v] + vbp + outbuf[gidx];          // ev_proj
      const float cand = tanhf(x);
      const float pv   = (t == 0) ? 0.0f : outbuf[gidx - D_DIM]; // belief[t-1]
      const float bel  = bf2f(leak[gidx]) * pv + bf2f(wrt[gidx]) * cand;
      outbuf[gidx] = bel;
      nxt[(size_t)b * D_DIM + col] = f2bf(bel);
    }

    grid_barrier(bar);
  }
}

// ---------------------------------------------------------------------------
extern "C" void kernel_launch(void* const* d_in, const int* in_sizes, int n_in,
                              void* d_out, int out_size, void* d_ws, size_t ws_size,
                              hipStream_t stream) {
  (void)in_sizes; (void)n_in; (void)out_size; (void)ws_size;

  const float* ev = (const float*)d_in[0];
  const float* We = (const float*)d_in[1];
  const float* be = (const float*)d_in[2];
  const float* Wp = (const float*)d_in[3];
  const float* bp = (const float*)d_in[4];
  const float* Wl = (const float*)d_in[5];
  const float* bl = (const float*)d_in[6];
  const float* Ww = (const float*)d_in[7];
  const float* bw = (const float*)d_in[8];
  float* out = (float*)d_out;

  // Workspace layout (bytes), total ~98.1 MB:
  //   0         WeT  bf16  524288
  //   524288    WlT  bf16  524288
  //   1048576   WwT  bf16  524288
  //   1572864   WpT  bf16  524288
  //   2097152   leak bf16  33554432
  //   35651584  wrt  bf16  33554432
  //   69206016  evbf bf16  33554432
  //   102760448 bel0 bf16  65536
  //   102825984 bel1 bf16  65536
  //   102891520 barrier (2 x u32)
  char* ws = (char*)d_ws;
  unsigned short* WeT  = (unsigned short*)(ws);
  unsigned short* WlT  = (unsigned short*)(ws + 524288);
  unsigned short* WwT  = (unsigned short*)(ws + 1048576);
  unsigned short* WpT  = (unsigned short*)(ws + 1572864);
  unsigned short* leak = (unsigned short*)(ws + 2097152);
  unsigned short* wrt  = (unsigned short*)(ws + 35651584);
  unsigned short* evbf = (unsigned short*)(ws + 69206016);
  unsigned short* bel0 = (unsigned short*)(ws + 102760448);
  unsigned short* bel1 = (unsigned short*)(ws + 102825984);
  unsigned int*   bar  = (unsigned int*)  (ws + 102891520);

  evconv_kernel<<<M_TOT * D_DIM / (256 * 4), 256, 0, stream>>>(ev, evbf);
  prep_kernel<<<1024, 256, 0, stream>>>(We, Wp, Wl, Ww, WeT, WpT, WlT, WwT, bel0, bar);

  proj_kernel<<<dim3(M_TOT / 16, D_DIM / 128), 256, 0, stream>>>(
      evbf, WeT, WlT, WwT, be, bl, bw, out, leak, wrt);

  scan_kernel<<<SCAN_BLOCKS, 256, 0, stream>>>(WpT, bp, leak, wrt, out, bel0, bel1, bar);
}
